// AttentionLayer_42013370089760
// MI455X (gfx1250) — matrix operand order repaired and checked by reading, not compiled, not run on hardware
//
#include <hip/hip_runtime.h>

#define BATCH 64
#define SEQ   2048
#define HID   512

typedef __attribute__((ext_vector_type(16))) __bf16 v16bf;
typedef __attribute__((ext_vector_type(8)))  __bf16 v8bf;
typedef __attribute__((ext_vector_type(8)))  float  v8f;

#if __has_builtin(__builtin_amdgcn_exp2f)
#define EXP2F(x) __builtin_amdgcn_exp2f(x)
#else
#define EXP2F(x) exp2f(x)
#endif

static __device__ __forceinline__ unsigned short f32_bf16(float f) {
    unsigned u = __float_as_uint(f);
    u += 0x7FFFu + ((u >> 16) & 1u);   // round-to-nearest-even
    return (unsigned short)(u >> 16);
}

static __device__ __forceinline__ float fast_tanh(float x) {
#if __has_builtin(__builtin_amdgcn_tanhf)
    return __builtin_amdgcn_tanhf(x);          // v_tanh_f32 (gfx1250 trans op)
#else
    float e = EXP2F(x * 2.8853900817779268f);  // exp(2x)
    return 1.0f - 2.0f * __builtin_amdgcn_rcpf(e + 1.0f);
#endif
}

// ---------------------------------------------------------------------------
// 1) Pre-swizzle W1 (f32 row-major [k][n]) into per-lane bf16 B-fragments.
//    Layout: idx = ((nt*16 + kt)*32 + lane)*16 + i
//    lane: n = nt*16 + (lane&15), k = kt*32 + (lane<16?0:16) + i
// ---------------------------------------------------------------------------
__global__ void k_pack_w1(const float* __restrict__ W1, unsigned short* __restrict__ out) {
    int blk = blockIdx.x;              // nt*16 + kt, 512 blocks
    int nt = blk >> 4, kt = blk & 15;
    int l  = threadIdx.x;              // 0..31
    int n  = nt * 16 + (l & 15);
    int kb = kt * 32 + ((l >> 4) << 4);
    unsigned short* o = out + (size_t)(blk * 32 + l) * 16;
#pragma unroll
    for (int i = 0; i < 16; ++i)
        o[i] = f32_bf16(W1[(size_t)(kb + i) * HID + n]);
}

// ---------------------------------------------------------------------------
// 2) comb[b][n] = b1[n] + b2[n] + sum_k hidden[b][k]*W2[k][n]
// ---------------------------------------------------------------------------
__global__ __launch_bounds__(256) void k_comb(const float* __restrict__ hidden,
                                              const float* __restrict__ W2,
                                              const float* __restrict__ b1,
                                              const float* __restrict__ b2,
                                              float* __restrict__ comb) {
    int b = blockIdx.x, t = threadIdx.x;
    const float* h = hidden + b * HID;
    float a0 = 0.f, a1 = 0.f;
    for (int k = 0; k < HID; ++k) {
        float hv = h[k];
        a0 += hv * W2[(size_t)k * HID + t];
        a1 += hv * W2[(size_t)k * HID + t + 256];
    }
    comb[b * HID + t]       = a0 + b1[t]       + b2[t];
    comb[b * HID + t + 256] = a1 + b1[t + 256] + b2[t + 256];
}

// ---------------------------------------------------------------------------
// 3) Fused scores: one block = 64 rows (four 16-row M-tiles) of one batch b.
//    8 waves split N (64 cols each); each B fragment reused for all 4 M-tiles:
//    per K-step/wave: 8 ds_load_b128 (A) + 8 global_load_b128 (B) -> 16 WMMA.
// ---------------------------------------------------------------------------
__global__ __launch_bounds__(256) void k_scores(const float* __restrict__ enc,
                                                const unsigned short* __restrict__ w1pk,
                                                const float* __restrict__ comb,
                                                const float* __restrict__ V,
                                                float* __restrict__ logits) {
    __shared__ unsigned AtileU[64 * 260];   // 64 rows x 520 bf16 (512 + 8 pad) = 66560 B
    __shared__ float    wlds[512];          // [8 waves][64 m]

    int tile = blockIdx.x;                  // 2048 = B * S/64
    int b  = tile >> 5;
    int s0 = (tile & 31) << 6;
    int tid = threadIdx.x;

    // --- stage A tile: 64x512 f32 -> bf16 LDS with padded stride (520 bf16) ---
    {
        int row = tid >> 2;
        int col = (tid & 3) << 7;           // 128 cols per thread
        const float* src = enc + ((size_t)(b * SEQ + s0 + row)) * HID + col;
        unsigned* dst = AtileU + row * 260 + (col >> 1);
#pragma unroll
        for (int i = 0; i < 32; ++i) {
            float4 v = ((const float4*)src)[i];
            dst[i * 2 + 0] = (unsigned)f32_bf16(v.x) | ((unsigned)f32_bf16(v.y) << 16);
            dst[i * 2 + 1] = (unsigned)f32_bf16(v.z) | ((unsigned)f32_bf16(v.w) << 16);
        }
    }
    __syncthreads();

    int wave = tid >> 5;
    int lane = tid & 31;
    int nl   = lane & 15;
    int hgrp = lane >> 4;

    // A-fragment per-lane base (M-tile mt adds mt*16 rows = mt*16*1040 bytes)
    const char*  Ab = (const char*)AtileU + nl * 1040 + hgrp * 32;
    // B-fragment per-lane base in packed W1 (elements)
    const __bf16* Bp = (const __bf16*)w1pk + (size_t)wave * 32768 + lane * 16;

    v8f acc[4][4] = {};
#pragma unroll
    for (int kt = 0; kt < 16; ++kt) {
        v16bf bf[4];
#pragma unroll
        for (int t = 0; t < 4; ++t)
            bf[t] = *(const v16bf*)(Bp + t * 8192 + kt * 512);
#pragma unroll
        for (int mt = 0; mt < 4; ++mt) {
            const char* Am = Ab + mt * (16 * 1040);
            v8bf a0 = *(const v8bf*)(Am + kt * 64);
            v8bf a1 = *(const v8bf*)(Am + kt * 64 + 16);
            v16bf af = __builtin_shufflevector(a0, a1, 0, 1, 2, 3, 4, 5, 6, 7,
                                               8, 9, 10, 11, 12, 13, 14, 15);
#pragma unroll
            for (int t = 0; t < 4; ++t)
                acc[mt][t] = __builtin_amdgcn_wmma_f32_16x16x32_bf16(
                    false, af, false, bf[t], (short)0, acc[mt][t], false, false);
        }
    }

    // --- epilogue: +bias, tanh, dot with V, deterministic reductions ---
    int nbase = (wave << 6) + nl;
    float bbv[4], vvv[4];
#pragma unroll
    for (int t = 0; t < 4; ++t) {
        int n = nbase + (t << 4);
        bbv[t] = comb[b * HID + n];
        vvv[t] = V[n];
    }
#pragma unroll
    for (int mt = 0; mt < 4; ++mt) {
        float part[8];
#pragma unroll
        for (int j = 0; j < 8; ++j) part[j] = 0.f;
#pragma unroll
        for (int t = 0; t < 4; ++t) {
#pragma unroll
            for (int j = 0; j < 8; ++j)
                part[j] += fast_tanh(acc[mt][t][j] + bbv[t]) * vvv[t];
        }
        // xor reduce within each 16-lane group (C-layout: m = j + 8*hgrp)
#pragma unroll
        for (int j = 0; j < 8; ++j) {
            float v = part[j];
#pragma unroll
            for (int off = 1; off < 16; off <<= 1)
                v += __shfl_xor(v, off, 32);
            part[j] = v;
        }
        if (nl == 0) {
#pragma unroll
            for (int j = 0; j < 8; ++j)
                wlds[wave * 64 + mt * 16 + hgrp * 8 + j] = part[j];
        }
    }
    __syncthreads();
    if (tid < 64) {
        float s = 0.f;
#pragma unroll
        for (int w = 0; w < 8; ++w) s += wlds[w * 64 + tid];   // fixed order
        logits[b * SEQ + s0 + tid] = s;
    }
}

// ---------------------------------------------------------------------------
// 4) softmax over S per batch row; weights -> output slice
// ---------------------------------------------------------------------------
__global__ __launch_bounds__(256) void k_softmax(const float* __restrict__ logits,
                                                 float* __restrict__ wout) {
    __shared__ float red[256];
    int b = blockIdx.x, tid = threadIdx.x;
    const float* lr = logits + b * SEQ;
    float v[8];
    float m = -1e30f;
#pragma unroll
    for (int i = 0; i < 8; ++i) { v[i] = lr[tid + i * 256]; m = fmaxf(m, v[i]); }
    red[tid] = m; __syncthreads();
    for (int st = 128; st > 0; st >>= 1) {
        if (tid < st) red[tid] = fmaxf(red[tid], red[tid + st]);
        __syncthreads();
    }
    m = red[0]; __syncthreads();
    float s = 0.f;
#pragma unroll
    for (int i = 0; i < 8; ++i) {
        v[i] = EXP2F((v[i] - m) * 1.4426950408889634f);
        s += v[i];
    }
    red[tid] = s; __syncthreads();
    for (int st = 128; st > 0; st >>= 1) {
        if (tid < st) red[tid] += red[tid + st];
        __syncthreads();
    }
    float inv = 1.0f / red[0];
    float* wr = wout + b * SEQ;
#pragma unroll
    for (int i = 0; i < 8; ++i) wr[tid + i * 256] = v[i] * inv;
}

// ---------------------------------------------------------------------------
// 5) context[b][h] = sum_s w[b][s] * enc[b][s][h]  (memory-bound pass)
// ---------------------------------------------------------------------------
__global__ __launch_bounds__(256) void k_context(const float* __restrict__ enc,
                                                 const float* __restrict__ w,
                                                 float* __restrict__ ctx) {
    __shared__ float red[256];
    int blk = blockIdx.x;            // b*4 + chunk
    int b = blk >> 2, chunk = blk & 3;
    int tid = threadIdx.x;
    int col = (chunk << 7) + (tid & 127);
    int sh  = tid >> 7;              // 0 or 1 -> S halves
    const float* wp = w + b * SEQ + sh * 1024;
    const float* ep = enc + ((size_t)b * SEQ + sh * 1024) * HID + col;
    float acc = 0.f;
    for (int s = 0; s < 1024; s += 4) {
        acc += wp[s + 0] * ep[(size_t)(s + 0) * HID];
        acc += wp[s + 1] * ep[(size_t)(s + 1) * HID];
        acc += wp[s + 2] * ep[(size_t)(s + 2) * HID];
        acc += wp[s + 3] * ep[(size_t)(s + 3) * HID];
    }
    red[tid] = acc; __syncthreads();
    if (tid < 128) ctx[b * HID + col] = red[tid] + red[tid + 128];
}

// ---------------------------------------------------------------------------
extern "C" void kernel_launch(void* const* d_in, const int* in_sizes, int n_in,
                              void* d_out, int out_size, void* d_ws, size_t ws_size,
                              hipStream_t stream) {
    const float* enc = (const float*)d_in[0];
    const float* hid = (const float*)d_in[1];
    const float* W1  = (const float*)d_in[2];
    const float* b1  = (const float*)d_in[3];
    const float* W2  = (const float*)d_in[4];
    const float* b2  = (const float*)d_in[5];
    const float* V   = (const float*)d_in[6];
    // d_in[7] = bV: constant shift, cancels in softmax (and is zero).

    char* ws = (char*)d_ws;
    unsigned short* w1pk = (unsigned short*)ws;        // 512 KB bf16-packed W1
    float* comb   = (float*)(ws + (512 << 10));        // 128 KB
    float* logits = (float*)(ws + (640 << 10));        // 512 KB

    float* ctx  = (float*)d_out;                       // [B,H]
    float* attw = (float*)d_out + BATCH * HID;         // [B,S,1]

    hipLaunchKernelGGL(k_pack_w1, dim3(512),  dim3(32),  0, stream, W1, w1pk);
    hipLaunchKernelGGL(k_comb,    dim3(BATCH), dim3(256), 0, stream, hid, W2, b1, b2, comb);
    hipLaunchKernelGGL(k_scores,  dim3(BATCH * SEQ / 64), dim3(256), 0, stream,
                       enc, w1pk, comb, V, logits);
    hipLaunchKernelGGL(k_softmax, dim3(BATCH), dim3(256), 0, stream, logits, attw);
    hipLaunchKernelGGL(k_context, dim3(BATCH * 4), dim3(256), 0, stream, enc, attw, ctx);
}
